// SelfAttention_30193620091497
// MI455X (gfx1250) — compile-verified, matrix operand order checked by reading, not run on hardware
//
#include <hip/hip_runtime.h>

// ---------------------------------------------------------------------------
// Problem constants (reference): x [8,32,64,512] fp32
//   tokens/batch N = 2048, feature dim 512, reduced dim 64
// ---------------------------------------------------------------------------
#define IN_DIM 512
#define RED    64
#define NTOK   2048
#define NB     8
#define MTOT   (NB * NTOK)   // 16384 rows

typedef __attribute__((ext_vector_type(16))) __bf16 v16bf;
typedef __attribute__((ext_vector_type(8)))  float  v8f;
typedef __attribute__((ext_vector_type(4)))  unsigned int u32x4;
typedef __attribute__((ext_vector_type(8)))  int i32x8;
typedef __attribute__((ext_vector_type(4)))  int i32x4;

__device__ __forceinline__ unsigned short f2bf(float f) {
  unsigned int u = __float_as_uint(f);
  unsigned int r = u + 0x7FFFu + ((u >> 16) & 1u);   // round-to-nearest-even
  return (unsigned short)(r >> 16);
}

// WMMA bf16 operand fragment: 16 bf16/lane (8 VGPRs), two 16B vectors matching
// the CDNA5 16-bit A/B layout (lanes 0-15: K 0-7 & 16-23; lanes 16-31: 8-15 & 24-31)
union Frag {
  uint4 u[2];
  v16bf v;
};

__device__ __forceinline__ Frag load_frag(const unsigned short* __restrict__ base,
                                          int row0, int k0, int ld) {
  int lane = threadIdx.x & 31;
  int r = lane & 15, hi = lane >> 4;
  const unsigned short* p = base + (size_t)(row0 + r) * ld + k0 + hi * 8;
  Frag f;
  f.u[0] = *(const uint4*)p;
  f.u[1] = *(const uint4*)(p + 16);
  return f;
}

__device__ __forceinline__ Frag load_frag_lds(const unsigned short* base,
                                              int row0, int ld) {
  int lane = threadIdx.x & 31;
  int r = lane & 15, hi = lane >> 4;
  const unsigned short* p = base + (row0 + r) * ld + hi * 8;
  Frag f;
  f.u[0] = *(const uint4*)p;        // ds_load_b128
  f.u[1] = *(const uint4*)(p + 16);
  return f;
}

__device__ __forceinline__ v8f wmma_bf16(const Frag& a, const Frag& b, v8f c) {
  return __builtin_amdgcn_wmma_f32_16x16x32_bf16(false, a.v, false, b.v,
                                                 (short)0, c, false, false);
}

// ---------------------------------------------------------------------------
// Tensor Data Mover: DMA a [rows x 32] bf16 tile (row stride = ld elements)
// from global memory into LDS.  D# built per CDNA5 ISA.
// Generic->LDS address: low 32 bits of a generic LDS pointer are the LDS
// byte offset (ISA 10.2 aperture mapping).
// ---------------------------------------------------------------------------
#if __has_builtin(__builtin_amdgcn_tensor_load_to_lds) && \
    __has_builtin(__builtin_amdgcn_s_wait_tensorcnt)
#define USE_TDM 1
#else
#define USE_TDM 0
#endif

__device__ __forceinline__ void tdm_load_tile(unsigned short* lds_dst,
                                              const unsigned short* gsrc,
                                              int rows, int ld) {
#if USE_TDM
  unsigned lds_addr = (unsigned)(unsigned long long)(size_t)lds_dst;
  unsigned long long ga = (unsigned long long)(size_t)gsrc;
  u32x4 g0;
  g0[0] = 1u;                                   // count=1 (valid user D#)
  g0[1] = lds_addr;                             // lds_addr
  g0[2] = (unsigned)(ga & 0xFFFFFFFFu);         // global_addr[31:0]
  g0[3] = (unsigned)((ga >> 32) & 0x1FFFFFFu)   // global_addr[56:32]
          | (2u << 30);                         // type=2 (image)
  unsigned td0 = (unsigned)ld;                  // tensor_dim0 (elements)
  unsigned td1 = (unsigned)rows;                // tensor_dim1
  i32x8 g1;
  g1[0] = (int)(1u << 16);                      // data_size=1 -> 2 bytes
  g1[1] = (int)((td0 & 0xFFFFu) << 16);         // tensor_dim0[15:0]
  g1[2] = (int)((td0 >> 16) | ((td1 & 0xFFFFu) << 16));
  g1[3] = (int)((td1 >> 16) | (32u << 16));     // tile_dim0 = 32
  g1[4] = (int)td1;                             // tile_dim1 = rows, tile_dim2 = 0
  g1[5] = (int)td0;                             // tensor_dim0_stride[31:0]
  g1[6] = 0;
  g1[7] = 0;
  i32x4 z4 = {0, 0, 0, 0};                      // 2-D tensor: groups 2/3 unused
#if defined(__clang_major__) && (__clang_major__ >= 23)
  i32x8 z8 = {0, 0, 0, 0, 0, 0, 0, 0};
  __builtin_amdgcn_tensor_load_to_lds(g0, g1, z4, z4, z8, 0);
#else
  __builtin_amdgcn_tensor_load_to_lds(g0, g1, z4, z4, 0);
#endif
#else
  // Fallback: cooperative copy (one uint4 per thread covers rows*32 bf16)
  int t = threadIdx.x;
  int total16 = rows * 32 / 8;                  // # of 16B chunks
  for (int c = t; c < total16; c += blockDim.x) {
    int row = c >> 2, chunk = c & 3;
    *(uint4*)(lds_dst + row * 32 + chunk * 8) =
        *(const uint4*)(gsrc + (size_t)row * ld + chunk * 8);
  }
#endif
}

__device__ __forceinline__ void tdm_wait_le1() {
#if USE_TDM
  __builtin_amdgcn_s_wait_tensorcnt(1);
#endif
}
__device__ __forceinline__ void tdm_wait_0() {
#if USE_TDM
  __builtin_amdgcn_s_wait_tensorcnt(0);
#endif
}

// ---------------------------------------------------------------------------
// fp32 -> bf16 conversions (x straight; weights transposed for the B-operand)
// ---------------------------------------------------------------------------
__global__ void cvt_f32_bf16_kernel(const float* __restrict__ in,
                                    unsigned short* __restrict__ outv, int n) {
  for (int i = blockIdx.x * blockDim.x + threadIdx.x; i < n;
       i += gridDim.x * blockDim.x)
    outv[i] = f2bf(in[i]);
}

__global__ void cvt_wT_kernel(const float* __restrict__ w,     // [in,out]
                              unsigned short* __restrict__ wT, // [out,in]
                              int in_dim, int out_dim) {
  int total = in_dim * out_dim;
  for (int idx = blockIdx.x * blockDim.x + threadIdx.x; idx < total;
       idx += gridDim.x * blockDim.x) {
    int o = idx % out_dim;
    int i = idx / out_dim;
    wT[(size_t)o * in_dim + i] = f2bf(w[(size_t)i * out_dim + o]);
  }
}

// ---------------------------------------------------------------------------
// bf16 GEMM: out[M,Nout] = A[M,K] * Bt[Nout,K]^T + bias (bf16 out).
// 256 threads = 8 waves, each wave owns a 32x64 tile (8 WMMA / k-step).
// The shared 64x32 B-tile is TDM-DMA'd into LDS, double-buffered so the
// tensor copy of k+32 overlaps WMMA on k (wave 0 drives the TDM).
// transposed==1 stores v transposed: vt[b][col][n].
// ---------------------------------------------------------------------------
__global__ void gemm_bf16_kernel(const unsigned short* __restrict__ A,
                                 const unsigned short* __restrict__ Bt,
                                 const float* __restrict__ bias,
                                 unsigned short* __restrict__ out,
                                 int Nout, int K, int transposed) {
  __shared__ unsigned short sB[2][64 * 32];     // 2 x 4KB B tiles
  int wave = threadIdx.x >> 5;
  int lane = threadIdx.x & 31;
  int m0 = blockIdx.x * 256 + wave * 32;
  int n0 = blockIdx.y * 64;
  const unsigned short* brow = Bt + (size_t)n0 * K;
  bool w0 = (threadIdx.x < 32);

  v8f acc[2][4];
#pragma unroll
  for (int i = 0; i < 2; ++i)
#pragma unroll
    for (int j = 0; j < 4; ++j) acc[i][j] = {};

#if USE_TDM
  if (w0) tdm_load_tile(&sB[0][0], brow, 64, K);
#else
  tdm_load_tile(&sB[0][0], brow, 64, K);
#endif

  int cur = 0;
  for (int k = 0; k < K; k += 32) {
    if (k + 32 < K) {
#if USE_TDM
      if (w0) { tdm_load_tile(&sB[cur ^ 1][0], brow + (k + 32), 64, K); tdm_wait_le1(); }
#else
      tdm_load_tile(&sB[cur ^ 1][0], brow + (k + 32), 64, K);
#endif
    } else {
      if (w0) tdm_wait_0();
    }
    __syncthreads();

    __builtin_prefetch(A + (size_t)m0 * K + k + 256, 0, 0);  // global_prefetch_b8
    Frag a0 = load_frag(A, m0, k, K);
    Frag a1 = load_frag(A, m0 + 16, k, K);
#pragma unroll
    for (int j = 0; j < 4; ++j) {
      Frag b = load_frag_lds(&sB[cur][0], j * 16, 32);
      acc[0][j] = wmma_bf16(a0, b, acc[0][j]);
      acc[1][j] = wmma_bf16(a1, b, acc[1][j]);
    }
    __syncthreads();            // readers done before buf is DMA'd again
    cur ^= 1;
  }

  int r15 = lane & 15, hi = lane >> 4;
#pragma unroll
  for (int i = 0; i < 2; ++i) {
#pragma unroll
    for (int j = 0; j < 4; ++j) {
#pragma unroll
      for (int r = 0; r < 8; ++r) {
        int row = m0 + i * 16 + r + hi * 8;   // C/D: lanes 16-31 -> M = r+8
        int col = n0 + j * 16 + r15;
        float v = acc[i][j][r] + bias[col];
        if (!transposed) {
          out[(size_t)row * Nout + col] = f2bf(v);
        } else {
          int bb = row >> 11;
          int n  = row & 2047;
          out[((size_t)bb * Nout + col) * NTOK + n] = f2bf(v);
        }
      }
    }
  }
}

// ---------------------------------------------------------------------------
// Pass 1: per-row softmax stats over energy = q k^T (never materialized).
// ---------------------------------------------------------------------------
__global__ void softmax_stats_kernel(const unsigned short* __restrict__ q,
                                     const unsigned short* __restrict__ kmat,
                                     float* __restrict__ sm,
                                     float* __restrict__ ss) {
  int wave = threadIdx.x >> 5;
  int lane = threadIdx.x & 31;
  int g0 = blockIdx.x * 128 + wave * 16;
  int b  = g0 >> 11;
  const unsigned short* kb = kmat + (size_t)b * NTOK * RED;

  Frag a0 = load_frag(q, g0, 0, RED);
  Frag a1 = load_frag(q, g0, 32, RED);

  float m[8], s[8];
#pragma unroll
  for (int r = 0; r < 8; ++r) { m[r] = -1e30f; s[r] = 0.0f; }

  for (int n0 = 0; n0 < NTOK; n0 += 16) {
    Frag b0 = load_frag(kb, n0, 0, RED);
    Frag b1 = load_frag(kb, n0, 32, RED);
    v8f c = {};
    c = wmma_bf16(a0, b0, c);
    c = wmma_bf16(a1, b1, c);
#pragma unroll
    for (int r = 0; r < 8; ++r) {
      float sv = c[r];
      float mn = fmaxf(m[r], sv);
      s[r] = s[r] * __expf(m[r] - mn) + __expf(sv - mn);
      m[r] = mn;
    }
  }
#pragma unroll
  for (int off = 1; off < 16; off <<= 1) {
#pragma unroll
    for (int r = 0; r < 8; ++r) {
      float om = __shfl_xor(m[r], off, 32);
      float os = __shfl_xor(s[r], off, 32);
      float mn = fmaxf(m[r], om);
      s[r] = s[r] * __expf(m[r] - mn) + os * __expf(om - mn);
      m[r] = mn;
    }
  }
  if ((lane & 15) == 0) {
    int hi = lane >> 4;
#pragma unroll
    for (int r = 0; r < 8; ++r) {
      int row = g0 + r + hi * 8;
      sm[row] = m[r];
      ss[row] = s[r];
    }
  }
}

// ---------------------------------------------------------------------------
// Pass 2: fused attention. WG = 16 query rows x full 512-dim output.
// Per 128-key block: each wave WMMAs one 16x16 S subtile, exp/scales it into
// a shared 16x128 bf16 P tile in LDS; then each wave WMMAs P x Vt for its
// private 64-dim slice (A-frags via ds_load_b128).
// ---------------------------------------------------------------------------
__global__ void attn_kernel(const unsigned short* __restrict__ q,
                            const unsigned short* __restrict__ kmat,
                            const unsigned short* __restrict__ vt, // [B,512,2048]
                            const float* __restrict__ sm,
                            const float* __restrict__ ss,
                            unsigned short* __restrict__ aout) {
  __shared__ unsigned short sP[16 * 128];
  int wave = threadIdx.x >> 5;
  int lane = threadIdx.x & 31;
  int r15 = lane & 15, hi = lane >> 4;
  int g0 = blockIdx.x * 16;
  int b  = g0 >> 11;
  const unsigned short* kb = kmat + (size_t)b * NTOK * RED;
  const unsigned short* vb = vt + (size_t)b * IN_DIM * NTOK;

  Frag a0 = load_frag(q, g0, 0, RED);
  Frag a1 = load_frag(q, g0, 32, RED);

  float mrow[8], irow[8];
#pragma unroll
  for (int r = 0; r < 8; ++r) {
    int row = g0 + r + hi * 8;
    mrow[r] = sm[row];
    irow[r] = 1.0f / ss[row];
  }

  int d0 = wave * 64;
  v8f acc[4];
#pragma unroll
  for (int j = 0; j < 4; ++j) acc[j] = {};

  for (int kbk = 0; kbk < NTOK; kbk += 128) {
    {
      int n0 = kbk + wave * 16;
      Frag b0 = load_frag(kb, n0, 0, RED);
      Frag b1 = load_frag(kb, n0, 32, RED);
      v8f c = {};
      c = wmma_bf16(a0, b0, c);
      c = wmma_bf16(a1, b1, c);
      int col = wave * 16 + r15;
#pragma unroll
      for (int r = 0; r < 8; ++r) {
        int row = r + hi * 8;
        float p = __expf(c[r] - mrow[r]) * irow[r];
        sP[row * 128 + col] = f2bf(p);
      }
    }
    __syncthreads();
#pragma unroll
    for (int kk = 0; kk < 128; kk += 32) {
      Frag ap;
      const unsigned short* p = sP + r15 * 128 + kk + hi * 8;
      ap.u[0] = *(const uint4*)p;
      ap.u[1] = *(const uint4*)(p + 16);
#pragma unroll
      for (int j = 0; j < 4; ++j) {
        Frag bv = load_frag(vb, d0 + j * 16, kbk + kk, NTOK);
        acc[j] = wmma_bf16(ap, bv, acc[j]);
      }
    }
    __syncthreads();
  }

#pragma unroll
  for (int j = 0; j < 4; ++j) {
#pragma unroll
    for (int r = 0; r < 8; ++r) {
      int row = g0 + r + hi * 8;
      int col = d0 + j * 16 + r15;
      aout[(size_t)row * IN_DIM + col] = f2bf(acc[j][r]);
    }
  }
}

// ---------------------------------------------------------------------------
// Output projection + bias + residual (fp32 out), TDM-staged woT tiles.
// ---------------------------------------------------------------------------
__global__ void final_proj_kernel(const unsigned short* __restrict__ A,
                                  const unsigned short* __restrict__ woT,
                                  const float* __restrict__ bo,
                                  const float* __restrict__ x,
                                  float* __restrict__ out) {
  __shared__ unsigned short sB[2][64 * 32];
  int wave = threadIdx.x >> 5;
  int lane = threadIdx.x & 31;
  int m0 = blockIdx.x * 256 + wave * 32;
  int n0 = blockIdx.y * 64;
  const unsigned short* brow = woT + (size_t)n0 * IN_DIM;
  bool w0 = (threadIdx.x < 32);

  v8f acc[2][4];
#pragma unroll
  for (int i = 0; i < 2; ++i)
#pragma unroll
    for (int j = 0; j < 4; ++j) acc[i][j] = {};

#if USE_TDM
  if (w0) tdm_load_tile(&sB[0][0], brow, 64, IN_DIM);
#else
  tdm_load_tile(&sB[0][0], brow, 64, IN_DIM);
#endif

  int cur = 0;
  for (int k = 0; k < IN_DIM; k += 32) {
    if (k + 32 < IN_DIM) {
#if USE_TDM
      if (w0) { tdm_load_tile(&sB[cur ^ 1][0], brow + (k + 32), 64, IN_DIM); tdm_wait_le1(); }
#else
      tdm_load_tile(&sB[cur ^ 1][0], brow + (k + 32), 64, IN_DIM);
#endif
    } else {
      if (w0) tdm_wait_0();
    }
    __syncthreads();

    __builtin_prefetch(A + (size_t)m0 * IN_DIM + k + 256, 0, 0);
    Frag a0 = load_frag(A, m0, k, IN_DIM);
    Frag a1 = load_frag(A, m0 + 16, k, IN_DIM);
#pragma unroll
    for (int j = 0; j < 4; ++j) {
      Frag b = load_frag_lds(&sB[cur][0], j * 16, 32);
      acc[0][j] = wmma_bf16(a0, b, acc[0][j]);
      acc[1][j] = wmma_bf16(a1, b, acc[1][j]);
    }
    __syncthreads();
    cur ^= 1;
  }

  int r15 = lane & 15, hi = lane >> 4;
#pragma unroll
  for (int i = 0; i < 2; ++i) {
#pragma unroll
    for (int j = 0; j < 4; ++j) {
#pragma unroll
      for (int r = 0; r < 8; ++r) {
        int row = m0 + i * 16 + r + hi * 8;
        int col = n0 + j * 16 + r15;
        out[(size_t)row * IN_DIM + col] =
            acc[i][j][r] + bo[col] + x[(size_t)row * IN_DIM + col];
      }
    }
  }
}

// ---------------------------------------------------------------------------
extern "C" void kernel_launch(void* const* d_in, const int* in_sizes, int n_in,
                              void* d_out, int out_size, void* d_ws, size_t ws_size,
                              hipStream_t stream) {
  (void)in_sizes; (void)n_in; (void)out_size; (void)ws_size;
  const float* x  = (const float*)d_in[0];
  const float* wq = (const float*)d_in[1];
  const float* bq = (const float*)d_in[2];
  const float* wk = (const float*)d_in[3];
  const float* bk = (const float*)d_in[4];
  const float* wv = (const float*)d_in[5];
  const float* bv = (const float*)d_in[6];
  const float* wo = (const float*)d_in[7];
  const float* bo = (const float*)d_in[8];
  float* out = (float*)d_out;

  char* ws = (char*)d_ws;
  size_t off = 0;
  auto alloc = [&](size_t bytes) -> void* {
    void* p = ws + off;
    off = (off + bytes + 255) & ~(size_t)255;
    return p;
  };
  unsigned short* xb   = (unsigned short*)alloc((size_t)MTOT * IN_DIM * 2);
  unsigned short* wqT  = (unsigned short*)alloc((size_t)RED * IN_DIM * 2);
  unsigned short* wkT  = (unsigned short*)alloc((size_t)RED * IN_DIM * 2);
  unsigned short* wvT  = (unsigned short*)alloc((size_t)IN_DIM * IN_DIM * 2);
  unsigned short* woT  = (unsigned short*)alloc((size_t)IN_DIM * IN_DIM * 2);
  unsigned short* qb   = (unsigned short*)alloc((size_t)MTOT * RED * 2);
  unsigned short* kbuf = (unsigned short*)alloc((size_t)MTOT * RED * 2);
  unsigned short* vtb  = (unsigned short*)alloc((size_t)NB * IN_DIM * NTOK * 2);
  float*          smv  = (float*)alloc((size_t)MTOT * 4);
  float*          ssv  = (float*)alloc((size_t)MTOT * 4);
  unsigned short* aob  = (unsigned short*)alloc((size_t)MTOT * IN_DIM * 2);

  dim3 blk(256);

  cvt_f32_bf16_kernel<<<2048, blk, 0, stream>>>(x, xb, MTOT * IN_DIM);
  cvt_wT_kernel<<<64,  blk, 0, stream>>>(wq, wqT, IN_DIM, RED);
  cvt_wT_kernel<<<64,  blk, 0, stream>>>(wk, wkT, IN_DIM, RED);
  cvt_wT_kernel<<<512, blk, 0, stream>>>(wv, wvT, IN_DIM, IN_DIM);
  cvt_wT_kernel<<<512, blk, 0, stream>>>(wo, woT, IN_DIM, IN_DIM);

  gemm_bf16_kernel<<<dim3(MTOT / 256, RED / 64), blk, 0, stream>>>(
      xb, wqT, bq, qb, RED, IN_DIM, 0);
  gemm_bf16_kernel<<<dim3(MTOT / 256, RED / 64), blk, 0, stream>>>(
      xb, wkT, bk, kbuf, RED, IN_DIM, 0);
  gemm_bf16_kernel<<<dim3(MTOT / 256, IN_DIM / 64), blk, 0, stream>>>(
      xb, wvT, bv, vtb, IN_DIM, IN_DIM, 1);

  softmax_stats_kernel<<<MTOT / 128, blk, 0, stream>>>(qb, kbuf, smv, ssv);
  attn_kernel<<<MTOT / 16, blk, 0, stream>>>(qb, kbuf, vtb, smv, ssv, aob);

  final_proj_kernel<<<dim3(MTOT / 256, IN_DIM / 64), blk, 0, stream>>>(
      aob, woT, bo, x, out);
}
